// VidHRFormerFAR_17523466568282
// MI455X (gfx1250) — matrix operand order repaired and checked by reading, not compiled, not run on hardware
//
// MI455X (gfx1250) implementation of the VidHRFormer-FAR forward pass.
// All dense projections (>98% of the ~316 GFLOP) run through
// v_wmma_f32_16x16x32_f16 (f16 in, fp32 accumulate). Weights are stored in
// fragment-swizzled f16 layout so B operands load straight from global
// (L2-resident) with per-lane 32B vector loads -- no LDS transpose.
// A (activations) is staged fp32->f16 through double-buffered LDS.
#include <hip/hip_runtime.h>
#include <cstddef>
#include <cstdint>

typedef __attribute__((ext_vector_type(16))) _Float16 v16h;
typedef __attribute__((ext_vector_type(8)))  float    v8f;

union HF16 { v16h v; _Float16 h[16]; float4 q[2]; };
union VF8  { v8f  v; float    f[8];  };

__device__ __forceinline__ float gelu_f(float x) {
  return 0.5f * x * (1.0f + erff(x * 0.7071067811865475f));
}

// A-tile staging: 16 elements per thread (one half-row of the 128x32 tile),
// packed into a v16h register then written as 2x ds_store_b128.
template<typename TA>
__device__ __forceinline__ void stage_tileA(const TA* __restrict__ ap,
                                            _Float16* adst) {
  if constexpr (sizeof(TA) == sizeof(float)) {
    const float4* p = reinterpret_cast<const float4*>(ap);
    float4 f0 = p[0], f1 = p[1], f2 = p[2], f3 = p[3];
    HF16 t;
    t.h[0]=(_Float16)f0.x;  t.h[1]=(_Float16)f0.y;
    t.h[2]=(_Float16)f0.z;  t.h[3]=(_Float16)f0.w;
    t.h[4]=(_Float16)f1.x;  t.h[5]=(_Float16)f1.y;
    t.h[6]=(_Float16)f1.z;  t.h[7]=(_Float16)f1.w;
    t.h[8]=(_Float16)f2.x;  t.h[9]=(_Float16)f2.y;
    t.h[10]=(_Float16)f2.z; t.h[11]=(_Float16)f2.w;
    t.h[12]=(_Float16)f3.x; t.h[13]=(_Float16)f3.y;
    t.h[14]=(_Float16)f3.z; t.h[15]=(_Float16)f3.w;
    *reinterpret_cast<v16h*>(adst) = t.v;
  } else {
    const float4* p = reinterpret_cast<const float4*>(ap);
    float4* d = reinterpret_cast<float4*>(adst);
    d[0] = p[0]; d[1] = p[1];
  }
}

// ---------------------------------------------------------------------------
// WMMA GEMM:  C[m,n] = epi( sum_k A[m,k]*B[k,n] + bias[n] )  (+res[m,n])
// Block = 256 threads = 8 wave32 waves; tile 128x128, K-step 32.
// Each wave: 32x64 tile = 2x4 grid of 16x16 v_wmma_f32_16x16x32_f16.
// A: fp32 (converted while staging) or f16, row-major (lda).
// B: f16 in fragment-swizzled layout:
//   element (k,n) at ((n/16)*KB + k/32)*512 + lane*16 + (k&15),
//   lane = (n&15) + 16*((k>>4)&1)   [contiguous k = 16*(lane/16) + e; A frags use the same order]
// EPI bit0 = exact GELU, bit1 = residual add. M,N multiples of 128.
// ---------------------------------------------------------------------------
template<typename TA, typename TO, int EPI>
__global__ __launch_bounds__(256) void gemm_wmma(
    const TA* __restrict__ A, int lda,
    const _Float16* __restrict__ B, int KB,   // KB = K/32
    const float* __restrict__ bias,
    const float* res, int ldres,
    TO* C, int ldc)
{
  constexpr int LAS = 40;   // A LDS row stride (halves) -> 80B, 16B aligned
  __shared__ _Float16 As[2][128 * LAS];

  const int tid  = threadIdx.x;
  const int lane = tid & 31;
  const int wave = tid >> 5;
  const int wr   = wave >> 1;      // 0..3 : 32-row band
  const int wc   = wave & 1;       // 0..1 : 64-col band
  const int row0 = blockIdx.y * 128;
  const int col0 = blockIdx.x * 128;
  const int nbBase = (col0 >> 4) + wc * 4;   // fragment column-block base

  v8f acc[2][4] = {};

  const int ar    = tid >> 1;          // 0..127
  const int acoff = (tid & 1) << 4;    // 0 / 16

  // prologue: stage first A tile
  stage_tileA(A + (size_t)(row0 + ar) * lda + acoff, &As[0][ar * LAS + acoff]);
  __syncthreads();

  const int lrow = lane & 15;
  const int kha  = (lane >> 4) << 4;   // A frag: lanes<16 K0..15, else 16..31

  for (int kb = 0; kb < KB; kb++) {
    const int cur = kb & 1;
    // stage next A tile into the other buffer (overlaps with this step's math)
    if (kb + 1 < KB) {
      stage_tileA(A + (size_t)(row0 + ar) * lda + ((kb + 1) * 32 + acoff),
                  &As[cur ^ 1][ar * LAS + acoff]);
      __builtin_prefetch(A + (size_t)(row0 + ar) * lda + ((kb + 2) * 32 + acoff), 0, 0);
      __builtin_prefetch(B + ((size_t)(nbBase * KB + kb + 1) * 32 + lane) * 16, 0, 0);
    }

    // ---- B fragments: direct per-lane 32B vector loads from global (L2) ----
    HF16 bf[4];
    #pragma unroll
    for (int ni = 0; ni < 4; ni++)
      bf[ni].v = *reinterpret_cast<const v16h*>(
          B + ((size_t)((nbBase + ni) * KB + kb) * 32 + lane) * 16);

    // ---- A fragments: contiguous ds_load_b128 pairs ----
    HF16 af[2];
    #pragma unroll
    for (int mi = 0; mi < 2; mi++) {
      const _Float16* s = &As[cur][(wr * 32 + mi * 16 + lrow) * LAS];
      af[mi].q[0] = *reinterpret_cast<const float4*>(s + kha);
      af[mi].q[1] = *reinterpret_cast<const float4*>(s + 8 + kha);
    }

    #pragma unroll
    for (int mi = 0; mi < 2; mi++)
      #pragma unroll
      for (int ni = 0; ni < 4; ni++)
        acc[mi][ni] = __builtin_amdgcn_wmma_f32_16x16x32_f16(
            false, af[mi].v, false, bf[ni].v, (short)0, acc[mi][ni], false, false);
    __syncthreads();
  }

  // ---- epilogue: bias / gelu / residual, per 32-bit C layout ----
  const int ncl = lane & 15;
  const int mo  = (lane >> 4) << 3;
  #pragma unroll
  for (int mi = 0; mi < 2; mi++) {
    #pragma unroll
    for (int ni = 0; ni < 4; ni++) {
      VF8 a; a.v = acc[mi][ni];
      const int gcol = col0 + wc * 64 + ni * 16 + ncl;
      const float bv = bias ? bias[gcol] : 0.0f;
      #pragma unroll
      for (int r = 0; r < 8; r++) {
        const int grow = row0 + wr * 32 + mi * 16 + mo + r;
        float v = a.f[r] + bv;
        if (EPI & 1) v = gelu_f(v);
        if (EPI & 2) v += res[(size_t)grow * ldres + gcol];
        C[(size_t)grow * ldc + gcol] = (TO)v;
      }
    }
  }
}

// ---------------------------------------------------------------------------
// LayerNorm over C=256, one wave32 per token.
// ---------------------------------------------------------------------------
__global__ __launch_bounds__(256) void layernorm_k(
    const float* __restrict__ x, const float* __restrict__ g,
    const float* __restrict__ b, float* __restrict__ y, int M)
{
  const int gw   = (blockIdx.x * blockDim.x + threadIdx.x) >> 5;
  const int lane = threadIdx.x & 31;
  if (gw >= M) return;
  const float* xr = x + (size_t)gw * 256;
  float v[8];
  float s = 0.f;
  #pragma unroll
  for (int e = 0; e < 8; e++) { v[e] = xr[lane * 8 + e]; s += v[e]; }
  #pragma unroll
  for (int o = 16; o; o >>= 1) s += __shfl_xor(s, o, 32);
  const float mu = s * (1.0f / 256.0f);
  float q = 0.f;
  #pragma unroll
  for (int e = 0; e < 8; e++) { float d = v[e] - mu; q += d * d; }
  #pragma unroll
  for (int o = 16; o; o >>= 1) q += __shfl_xor(q, o, 32);
  const float inv = rsqrtf(q * (1.0f / 256.0f) + 1e-5f);
  float* yr = y + (size_t)gw * 256;
  #pragma unroll
  for (int e = 0; e < 8; e++) {
    const int c = lane * 8 + e;
    yr[c] = (v[e] - mu) * inv * g[c] + b[c];
  }
}

// final encoder LN + ReLU + permute (N,T,H,W,C)->(N,T,C,H,W)
__global__ __launch_bounds__(256) void final_out_k(
    const float* __restrict__ x, const float* __restrict__ g,
    const float* __restrict__ b, float* __restrict__ out, int M)
{
  const int gw   = (blockIdx.x * blockDim.x + threadIdx.x) >> 5;
  const int lane = threadIdx.x & 31;
  if (gw >= M) return;
  const float* xr = x + (size_t)gw * 256;
  float v[8];
  float s = 0.f;
  #pragma unroll
  for (int e = 0; e < 8; e++) { v[e] = xr[lane * 8 + e]; s += v[e]; }
  #pragma unroll
  for (int o = 16; o; o >>= 1) s += __shfl_xor(s, o, 32);
  const float mu = s * (1.0f / 256.0f);
  float q = 0.f;
  #pragma unroll
  for (int e = 0; e < 8; e++) { float d = v[e] - mu; q += d * d; }
  #pragma unroll
  for (int o = 16; o; o >>= 1) q += __shfl_xor(q, o, 32);
  const float inv = rsqrtf(q * (1.0f / 256.0f) + 1e-5f);
  const int bt = gw / 784;
  const int hw = gw - bt * 784;
  float* ob = out + (size_t)bt * 256 * 784 + hw;
  #pragma unroll
  for (int e = 0; e < 8; e++) {
    const int c = lane * 8 + e;
    float val = (v[e] - mu) * inv * g[c] + b[c];
    ob[(size_t)c * 784] = fmaxf(val, 0.0f);
  }
}

// (N,T,C,H,W) -> token-major (N*T*H*W, C)
__global__ __launch_bounds__(256) void to_tokens_k(
    const float* __restrict__ in, float* __restrict__ x, int total)
{
  const int i = blockIdx.x * 256 + threadIdx.x;
  if (i >= total) return;
  const int hw = i % 784;
  const int r  = i / 784;
  const int c  = r % 256;
  const int bt = r / 256;
  x[((size_t)bt * 784 + hw) * 256 + c] = in[i];
}

// y + window position embedding (depends on h%7, w%7)
__global__ __launch_bounds__(256) void add_wpos_k(
    const float* __restrict__ y, const float* __restrict__ pos,
    float* __restrict__ o, int total)
{
  const int i = blockIdx.x * 256 + threadIdx.x;
  if (i >= total) return;
  const int c   = i & 255;
  const int tok = i >> 8;
  const int hw  = tok % 784;
  const int h   = hw / 28;
  const int w   = hw - h * 28;
  o[i] = y[i] + pos[((h % 7) * 7 + (w % 7)) * 256 + c];
}

// y + temporal position embedding (depends on t)
__global__ __launch_bounds__(256) void add_tpe_k(
    const float* __restrict__ y, const float* __restrict__ tpe,
    float* __restrict__ o, int total)
{
  const int i = blockIdx.x * 256 + threadIdx.x;
  if (i >= total) return;
  const int c   = i & 255;
  const int tok = i >> 8;
  const int t   = (tok / 784) & 15;
  o[i] = y[i] + tpe[t * 256 + c];
}

// depthwise 3x3 SAME conv over (NT,28,28,1024) f16 + bias + GELU
__global__ __launch_bounds__(256) void dwconv_k(
    const _Float16* __restrict__ hin, const float* __restrict__ w,
    const float* __restrict__ bias, _Float16* __restrict__ hout, int total)
{
  const int i = blockIdx.x * 256 + threadIdx.x;
  if (i >= total) return;
  const int c   = i & 1023;
  const int tok = i >> 10;
  const int b   = tok / 784;
  const int hw  = tok - b * 784;
  const int y   = hw / 28;
  const int x   = hw - y * 28;
  float s = 0.f;
  #pragma unroll
  for (int dy = -1; dy <= 1; dy++) {
    const int yy = y + dy;
    if ((unsigned)yy >= 28u) continue;
    #pragma unroll
    for (int dx = -1; dx <= 1; dx++) {
      const int xx = x + dx;
      if ((unsigned)xx >= 28u) continue;
      s += (float)hin[((size_t)b * 784 + yy * 28 + xx) * 1024 + c] *
           w[((dy + 1) * 3 + (dx + 1)) * 1024 + c];
    }
  }
  hout[i] = (_Float16)gelu_f(s + bias[c]);
}

// ---------------------------------------------------------------------------
// Window attention: one block (64 threads) per (window, head).
// QKV buffer layout per token (ld=768): [Q 0..255 | K 256..511 | V 512..767].
// Relative-position bias computed directly: idx = (dr+6)*13 + (dc+6).
// ---------------------------------------------------------------------------
__global__ __launch_bounds__(64) void win_attn_k(
    const float* __restrict__ qkv, const float* __restrict__ rpb,
    float* __restrict__ out)
{
  const int head = blockIdx.x & 7;
  const int win  = blockIdx.x >> 3;
  const int b    = win >> 4;           // NT image index (16 windows each)
  const int wi   = (win >> 2) & 3;
  const int wj   = win & 3;

  __shared__ float Q[49][32], Ks[49][32], V[49][32];
  __shared__ int   toks[49];
  const int tid = threadIdx.x;
  if (tid < 49) {
    const int r = tid / 7, s = tid % 7;
    toks[tid] = (b * 28 + wi * 7 + r) * 28 + (wj * 7 + s);
  }
  __syncthreads();
  for (int i = tid; i < 49 * 32; i += 64) {
    const int p = i >> 5, d = i & 31;
    const float* base = qkv + (size_t)toks[p] * 768 + head * 32 + d;
    Q[p][d]  = base[0];
    Ks[p][d] = base[256];
    V[p][d]  = base[512];
  }
  __syncthreads();
  if (tid >= 49) return;
  const int p  = tid;
  const int pr = p / 7, pc = p % 7;
  float logit[49];
  float mx = -1e30f;
  for (int q = 0; q < 49; q++) {
    float s = 0.f;
    #pragma unroll
    for (int d = 0; d < 32; d++) s += Q[p][d] * Ks[q][d];
    s *= 0.1767766953f;  // 1/sqrt(32)
    const int dr = pr - q / 7 + 6;
    const int dc = pc - q % 7 + 6;
    s += rpb[(dr * 13 + dc) * 8 + head];
    logit[q] = s;
    mx = fmaxf(mx, s);
  }
  float den = 0.f;
  for (int q = 0; q < 49; q++) { logit[q] = __expf(logit[q] - mx); den += logit[q]; }
  const float inv = 1.0f / den;
  float* ob = out + (size_t)toks[p] * 256 + head * 32;
  for (int d = 0; d < 32; d++) {
    float o = 0.f;
    for (int q = 0; q < 49; q++) o += logit[q] * V[q][d];
    ob[d] = o * inv;
  }
}

// ---------------------------------------------------------------------------
// Temporal causal attention over T=16: one wave per (sequence, head);
// lanes 0..15 each own one query timestep.
// ---------------------------------------------------------------------------
__global__ __launch_bounds__(256) void temp_attn_k(
    const float* __restrict__ qkv, float* __restrict__ out, int nwork)
{
  const int gw   = (blockIdx.x * 256 + threadIdx.x) >> 5;
  const int lane = threadIdx.x & 31;
  if (gw >= nwork || lane >= 16) return;
  const int head = gw & 7;
  const int seq  = gw >> 3;           // 0..1567 = n*784 + hw
  const int n    = seq / 784;
  const int hw   = seq - n * 784;
  const int t    = lane;
  const size_t tok_t = (size_t)(n * 16 + t) * 784 + hw;
  const float* qp = qkv + tok_t * 768 + head * 32;
  float q[32];
  #pragma unroll
  for (int d = 0; d < 32; d++) q[d] = qp[d];
  float logit[16];
  float mx = -1e30f;
  for (int k = 0; k <= t; k++) {
    const float* kp = qkv + ((size_t)(n * 16 + k) * 784 + hw) * 768 + 256 + head * 32;
    float s = 0.f;
    #pragma unroll
    for (int d = 0; d < 32; d++) s += q[d] * kp[d];
    s *= 0.1767766953f;
    logit[k] = s;
    mx = fmaxf(mx, s);
  }
  float den = 0.f;
  for (int k = 0; k <= t; k++) { logit[k] = __expf(logit[k] - mx); den += logit[k]; }
  const float inv = 1.0f / den;
  float* ob = out + tok_t * 256 + head * 32;
  for (int d = 0; d < 32; d++) {
    float o = 0.f;
    for (int k = 0; k <= t; k++)
      o += logit[k] * qkv[((size_t)(n * 16 + k) * 784 + hw) * 768 + 512 + head * 32 + d];
    ob[d] = o * inv;
  }
}

// ---------------------------------------------------------------------------
// Weight conversion into the fragment-swizzled f16 layout:
//   dst(((n>>4)*KB + (k>>5))*512 + ((n&15) + 16*((k>>4)&1))*16 + (k&15))
// ---------------------------------------------------------------------------
__device__ __forceinline__ size_t sw_idx(int k, int n, int KB) {
  const int lane = (n & 15) + ((k & 16) ? 16 : 0);
  return ((size_t)((n >> 4) * KB + (k >> 5)) * 32 + lane) * 16 + (k & 15);
}
// src W[d][c] (R rows = out dim d = n, Cc cols = in dim c = k)
__global__ __launch_bounds__(256) void cvt_nt_sw_k(
    const float* __restrict__ w, _Float16* __restrict__ o, int R, int Cc)
{
  const int i = blockIdx.x * 256 + threadIdx.x;
  if (i >= R * Cc) return;
  const int n = i / Cc;
  const int k = i - n * Cc;
  o[sw_idx(k, n, Cc >> 5)] = (_Float16)w[i];
}
// src W[c][d] (K rows = in dim = k, N cols = out dim = n)
__global__ __launch_bounds__(256) void cvt_kn_sw_k(
    const float* __restrict__ w, _Float16* __restrict__ o, int K, int N)
{
  const int i = blockIdx.x * 256 + threadIdx.x;
  if (i >= K * N) return;
  const int k = i / N;
  const int n = i - k * N;
  o[sw_idx(k, n, K >> 5)] = (_Float16)w[i];
}

// ---------------------------------------------------------------------------
extern "C" void kernel_launch(void* const* d_in, const int* in_sizes, int n_in,
                              void* d_out, int out_size, void* d_ws, size_t ws_size,
                              hipStream_t stream)
{
  (void)in_sizes; (void)n_in; (void)out_size; (void)ws_size;
  constexpr int L = 4, M = 25088;                 // 2*16*28*28 tokens
  constexpr int TOT = M * 256;                    // elements of a (M,256) tensor

  const float* input_feat = (const float*)d_in[0];
  const float* wpos       = (const float*)d_in[1];
  const float* tpe        = (const float*)d_in[2];
  const float* ln1_g = (const float*)d_in[3];  const float* ln1_b = (const float*)d_in[4];
  const float* sa_qkv_w = (const float*)d_in[5]; const float* sa_qkv_b = (const float*)d_in[6];
  const float* sa_out_w = (const float*)d_in[7]; const float* sa_out_b = (const float*)d_in[8];
  const float* rpb_tab  = (const float*)d_in[9];
  const float* ln2_g = (const float*)d_in[10]; const float* ln2_b = (const float*)d_in[11];
  const float* fc1_w = (const float*)d_in[12]; const float* fc1_b = (const float*)d_in[13];
  const float* dw_w  = (const float*)d_in[14]; const float* dw_b  = (const float*)d_in[15];
  const float* fc2_w = (const float*)d_in[16]; const float* fc2_b = (const float*)d_in[17];
  const float* ln3_g = (const float*)d_in[18]; const float* ln3_b = (const float*)d_in[19];
  const float* t_qkv_w = (const float*)d_in[20]; const float* t_qkv_b = (const float*)d_in[21];
  const float* t_out_w = (const float*)d_in[22]; const float* t_out_b = (const float*)d_in[23];
  const float* ln4_g = (const float*)d_in[24]; const float* ln4_b = (const float*)d_in[25];
  const float* lin1_w = (const float*)d_in[26]; const float* lin1_b = (const float*)d_in[27];
  const float* lin2_w = (const float*)d_in[28]; const float* lin2_b = (const float*)d_in[29];
  const float* enc_g = (const float*)d_in[30]; const float* enc_b = (const float*)d_in[31];

  // ---- workspace carve-up (all 256B aligned) ----
  char* ws = (char*)d_ws;
  size_t off = 0;
  auto alloc = [&](size_t bytes) -> void* {
    void* p = ws + off;
    off = (off + bytes + 255) & ~(size_t)255;
    return p;
  };
  float*     X   = (float*)alloc((size_t)TOT * 4);        // residual stream
  float*     T0  = (float*)alloc((size_t)TOT * 4);        // LN output
  float*     T1  = (float*)alloc((size_t)TOT * 4);        // LN + pos-embed
  float*     ATT = (float*)alloc((size_t)TOT * 4);        // attention output
  float*     QKV = (float*)alloc((size_t)M * 768 * 4);    // packed Q|K|V
  _Float16*  H1  = (_Float16*)alloc((size_t)M * 1024 * 2);
  _Float16*  H2  = (_Float16*)alloc((size_t)M * 1024 * 2);
  _Float16*  WH  = (_Float16*)alloc((size_t)L * 1572864 * 2); // swizzled f16 weights

  const dim3 blk(256);
  const int gEW = (TOT + 255) / 256;            // elementwise over (M,256)
  const int gLN = (M * 32 + 255) / 256;         // one wave per token
  const dim3 gN256(2, 196), gN512(4, 196), gN1024(8, 196);

  // per-layer f16 weight block offsets (halves)
  //   qkvT 0 | outT 196608 | fc1 262144 | fc2 524288
  //   tqkvT 786432 | toutT 983040 | lin1 1048576 | lin2 1310720
  // V sub-block inside a qkvT block starts at n0*K = 512*256 = 131072.
  for (int l = 0; l < L; l++) {
    _Float16* wl = WH + (size_t)l * 1572864;
    cvt_nt_sw_k<<<(768*256+255)/256, blk, 0, stream>>>(sa_qkv_w + (size_t)l*768*256, wl + 0,       768, 256);
    cvt_nt_sw_k<<<(256*256+255)/256, blk, 0, stream>>>(sa_out_w + (size_t)l*256*256, wl + 196608,  256, 256);
    cvt_kn_sw_k<<<(262144+255)/256,  blk, 0, stream>>>(fc1_w    + (size_t)l*262144,  wl + 262144,  256, 1024);
    cvt_kn_sw_k<<<(262144+255)/256,  blk, 0, stream>>>(fc2_w    + (size_t)l*262144,  wl + 524288,  1024, 256);
    cvt_nt_sw_k<<<(768*256+255)/256, blk, 0, stream>>>(t_qkv_w  + (size_t)l*768*256, wl + 786432,  768, 256);
    cvt_nt_sw_k<<<(256*256+255)/256, blk, 0, stream>>>(t_out_w  + (size_t)l*256*256, wl + 983040,  256, 256);
    cvt_kn_sw_k<<<(262144+255)/256,  blk, 0, stream>>>(lin1_w   + (size_t)l*262144,  wl + 1048576, 256, 1024);
    cvt_kn_sw_k<<<(262144+255)/256,  blk, 0, stream>>>(lin2_w   + (size_t)l*262144,  wl + 1310720, 1024, 256);
  }

  // ---- (N,T,C,H,W) -> token-major residual stream ----
  to_tokens_k<<<gEW, blk, 0, stream>>>(input_feat, X, TOT);

  for (int l = 0; l < L; l++) {
    _Float16* wl = WH + (size_t)l * 1572864;
    const float* qkvb  = sa_qkv_b + (size_t)l * 768;
    const float* tqkvb = t_qkv_b  + (size_t)l * 768;

    // ===== spatial window attention =====
    layernorm_k<<<gLN, blk, 0, stream>>>(X, ln1_g + l*256, ln1_b + l*256, T0, M);
    add_wpos_k <<<gEW, blk, 0, stream>>>(T0, wpos, T1, TOT);
    // Q,K from (y+pos); V from y
    gemm_wmma<float, float, 0><<<gN512, blk, 0, stream>>>(
        T1, 256, wl + 0, 8, qkvb, nullptr, 0, QKV, 768);
    gemm_wmma<float, float, 0><<<gN256, blk, 0, stream>>>(
        T0, 256, wl + 131072, 8, qkvb + 512, nullptr, 0, QKV + 512, 768);
    win_attn_k<<<512 * 8, dim3(64), 0, stream>>>(QKV, rpb_tab + (size_t)l*169*8, ATT);
    gemm_wmma<float, float, 2><<<gN256, blk, 0, stream>>>(
        ATT, 256, wl + 196608, 8, sa_out_b + l*256, X, 256, X, 256);

    // ===== spatial conv FFN =====
    layernorm_k<<<gLN, blk, 0, stream>>>(X, ln2_g + l*256, ln2_b + l*256, T0, M);
    gemm_wmma<float, _Float16, 1><<<gN1024, blk, 0, stream>>>(
        T0, 256, wl + 262144, 8, fc1_b + l*1024, nullptr, 0, H1, 1024);
    dwconv_k<<<(M*1024+255)/256, blk, 0, stream>>>(
        H1, dw_w + (size_t)l*9*1024, dw_b + l*1024, H2, M*1024);
    gemm_wmma<_Float16, float, 2><<<gN256, blk, 0, stream>>>(
        H2, 1024, wl + 524288, 32, fc2_b + l*256, X, 256, X, 256);

    // ===== temporal causal attention =====
    layernorm_k<<<gLN, blk, 0, stream>>>(X, ln3_g + l*256, ln3_b + l*256, T0, M);
    add_tpe_k  <<<gEW, blk, 0, stream>>>(T0, tpe, T1, TOT);
    gemm_wmma<float, float, 0><<<gN512, blk, 0, stream>>>(
        T1, 256, wl + 786432, 8, tqkvb, nullptr, 0, QKV, 768);
    gemm_wmma<float, float, 0><<<gN256, blk, 0, stream>>>(
        T0, 256, wl + 786432 + 131072, 8, tqkvb + 512, nullptr, 0, QKV + 512, 768);
    temp_attn_k<<<(1568*8*32 + 255)/256, blk, 0, stream>>>(QKV, ATT, 1568*8);
    gemm_wmma<float, float, 2><<<gN256, blk, 0, stream>>>(
        ATT, 256, wl + 983040, 8, t_out_b + l*256, X, 256, X, 256);

    // ===== temporal FFN =====
    layernorm_k<<<gLN, blk, 0, stream>>>(X, ln4_g + l*256, ln4_b + l*256, T0, M);
    gemm_wmma<float, _Float16, 1><<<gN1024, blk, 0, stream>>>(
        T0, 256, wl + 1048576, 8, lin1_b + l*1024, nullptr, 0, H1, 1024);
    gemm_wmma<_Float16, float, 2><<<gN256, blk, 0, stream>>>(
        H1, 1024, wl + 1310720, 32, lin2_b + l*256, X, 256, X, 256);
  }

  // ---- final LN + ReLU + permute to (N,T,C,H,W) ----
  final_out_k<<<gLN, blk, 0, stream>>>(X, enc_g, enc_b, (float*)d_out, M);
}
